// GNN_14216341750150
// MI455X (gfx1250) — matrix operand order repaired and checked by reading, not compiled, not run on hardware
//
#include <hip/hip_runtime.h>

// ---------------------------------------------------------------------------
// MI455X (gfx1250) GraphNet forward.
// Edge/Node MLPs run as wave-level 16xK @ Kx64 GEMMs on v_wmma_f32_16x16x32_bf16
// (fp32 accumulate).  Scatter reductions use fp32 atomics.  wave32 everywhere.
// ---------------------------------------------------------------------------

typedef __bf16 bf16;
typedef __attribute__((ext_vector_type(8)))  __bf16 v8bf;
typedef __attribute__((ext_vector_type(16))) __bf16 v16bf;
typedef __attribute__((ext_vector_type(8)))  float  v8f;

#define KE_MAX 192   // max wmma-K for edge MLP (2*64+64)
#define KN_MAX 256   // max wmma-K for node MLP (64+3*64), u handled as rank-1 term

static __device__ __forceinline__ v16bf ld_frag16(const bf16* p0, const bf16* p1) {
  v8bf lo = *(const v8bf*)p0;
  v8bf hi = *(const v8bf*)p1;
  return __builtin_shufflevector(lo, hi, 0,1,2,3,4,5,6,7,8,9,10,11,12,13,14,15);
}

// D += X[16][Kpad] @ W[Kpad][64], W staged transposed in LDS as Wt[64][Kpad].
// A layout (16-bit 16x32): lane m (0..15) & m+16 hold row M=m; K groups 0..7 /
// 8..15 selected by half, elements 8..15 are K+16.  B: lane&15 = N, same K order
// as A (K 8*half+0..7, then K+16).  C: M = v + 8*half, N = lane&15.
static __device__ __forceinline__ void wave_gemm16(const bf16* X, const bf16* Wt,
                                                   int Kpad, int lane, v8f acc[4]) {
  const int m = lane & 15;
  const int half = lane >> 4;
  for (int k0 = 0; k0 < Kpad; k0 += 32) {
    const bf16* xa = X + m * Kpad + k0 + 8 * half;
    v16bf a = ld_frag16(xa, xa + 16);
#pragma unroll
    for (int c = 0; c < 4; ++c) {
      const bf16* wb = Wt + (16 * c + m) * Kpad + k0 + 8 * half;
      v16bf b = ld_frag16(wb, wb + 16);
      acc[c] = __builtin_amdgcn_wmma_f32_16x16x32_bf16(
          /*neg_a=*/false, a, /*neg_b=*/false, b,
          /*c_mod=*/(short)0, acc[c], /*reuse_a=*/false, /*reuse_b=*/false);
    }
  }
}

static __device__ __forceinline__ float gelu_exact(float x) {
  return 0.5f * x * (1.0f + erff(x * 0.7071067811865475f));
}

static __device__ __forceinline__ void atomicMaxF(float* a, float v) {
  if (v >= 0.0f) atomicMax((int*)a, __float_as_int(v));
  else           atomicMin((unsigned int*)a, __float_as_uint(v));
}

// ---------------------------------------------------------------------------
// Edge MLP: e_out = LN(GELU([h_row|h_col|ea] W1 + b1) W2 + b2)*g + be (+ea)
// 64 threads = 2 waves, each wave owns a 16-edge tile.
// ---------------------------------------------------------------------------
__global__ __launch_bounds__(64) void edge_mlp_kernel(
    const float* __restrict__ h, int h_stride,
    const int* __restrict__ row, const int* __restrict__ col,
    const float* __restrict__ ea_in, int ea_stride,
    float* __restrict__ ea_out,
    const float* __restrict__ W1, const float* __restrict__ W2,
    const float* __restrict__ b1, const float* __restrict__ b2,
    const float* __restrict__ gg, const float* __restrict__ be,
    int E, int nin, int Kin, int Kpad, int residual)
{
  __shared__ alignas(16) bf16 sW1t[64 * KE_MAX];
  __shared__ alignas(16) bf16 sW2t[64 * 64];
  __shared__ alignas(16) bf16 sX[2][16 * KE_MAX];
  __shared__ alignas(16) bf16 sH[2][16 * 64];
  __shared__ float sY[2][16 * 64];
  __shared__ float sb1[64], sb2[64], sg[64], sbe[64];

  const int tid = threadIdx.x;
  if (tid == 0) { __builtin_prefetch(W1, 0, 1); __builtin_prefetch(W2, 0, 1); }

  for (int i = tid; i < 64 * Kpad; i += 64) {           // W1 transposed, zero-padded K
    int n = i / Kpad, k = i - n * Kpad;
    sW1t[i] = (k < Kin) ? (bf16)W1[k * 64 + n] : (bf16)0.0f;
  }
  for (int i = tid; i < 64 * 64; i += 64) {             // W2 transposed
    int n = i >> 6, k = i & 63;
    sW2t[i] = (bf16)W2[k * 64 + n];
  }
  sb1[tid] = b1[tid]; sb2[tid] = b2[tid]; sg[tid] = gg[tid]; sbe[tid] = be[tid];

  const int wave = tid >> 5, lane = tid & 31;
  const int e0 = (blockIdx.x * 2 + wave) * 16;
  bf16* X = sX[wave];

  // gather [h[row] | h[col] | edge_attr] into bf16 tile, zero-padded to Kpad
  for (int i = lane; i < 16 * Kpad; i += 32) {
    int r = i / Kpad, k = i - r * Kpad;
    int e = e0 + r;
    float v = 0.0f;
    if (e < E) {
      if (k < nin)          v = h[(long)row[e] * h_stride + k];
      else if (k < 2 * nin) v = h[(long)col[e] * h_stride + (k - nin)];
      else if (k < Kin)     v = ea_in[(long)e * ea_stride + (k - 2 * nin)];
    }
    X[i] = (bf16)v;
  }
  __syncthreads();

  v8f acc[4];
#pragma unroll
  for (int c = 0; c < 4; ++c)
#pragma unroll
    for (int v = 0; v < 8; ++v) acc[c][v] = 0.0f;
  wave_gemm16(X, sW1t, Kpad, lane, acc);

  { // h1 = gelu(acc + b1) -> bf16 tile for second GEMM
    const int m = lane & 15, half = lane >> 4;
#pragma unroll
    for (int c = 0; c < 4; ++c)
#pragma unroll
      for (int v = 0; v < 8; ++v) {
        int rowM = v + 8 * half, colf = 16 * c + m;
        sH[wave][rowM * 64 + colf] = (bf16)gelu_exact(acc[c][v] + sb1[colf]);
      }
  }
  __syncthreads();

  v8f acc2[4];
#pragma unroll
  for (int c = 0; c < 4; ++c)
#pragma unroll
    for (int v = 0; v < 8; ++v) acc2[c][v] = 0.0f;
  wave_gemm16(sH[wave], sW2t, 64, lane, acc2);

  {
    const int m = lane & 15, half = lane >> 4;
#pragma unroll
    for (int c = 0; c < 4; ++c)
#pragma unroll
      for (int v = 0; v < 8; ++v)
        sY[wave][(v + 8 * half) * 64 + 16 * c + m] = acc2[c][v] + sb2[16 * c + m];
  }
  __syncthreads();

  if (lane < 16) {                                     // layernorm + residual
    int e = e0 + lane;
    if (e < E) {
      const float* y = &sY[wave][lane * 64];
      float mean = 0.0f;
      for (int f = 0; f < 64; ++f) mean += y[f];
      mean *= (1.0f / 64.0f);
      float var = 0.0f;
      for (int f = 0; f < 64; ++f) { float d = y[f] - mean; var += d * d; }
      float inv = rsqrtf(var * (1.0f / 64.0f) + 1e-5f);
      float* o = ea_out + (long)e * 64;
      for (int f = 0; f < 64; ++f) {
        float ln = (y[f] - mean) * inv * sg[f] + sbe[f];
        o[f] = residual ? ln + ea_in[(long)e * ea_stride + f] : ln;
      }
    }
  }
}

// ---------------------------------------------------------------------------
// Scatter: segment sum / max / count of edge features over destination nodes.
// ---------------------------------------------------------------------------
__global__ void edge_scatter_kernel(const float* __restrict__ ea,
                                    const int* __restrict__ col,
                                    float* __restrict__ s, float* __restrict__ mx,
                                    float* __restrict__ cnt, int E) {
  long idx = (long)blockIdx.x * blockDim.x + threadIdx.x;
  long stride = (long)gridDim.x * blockDim.x;
  long tot = (long)E * 64;
  for (; idx < tot; idx += stride) {
    int e = (int)(idx >> 6), f = (int)(idx & 63);
    int c = col[e];
    float v = ea[idx];
    atomicAdd(&s[(long)c * 64 + f], v);
    atomicMaxF(&mx[(long)c * 64 + f], v);
    if (f == 0) atomicAdd(&cnt[c], 1.0f);
  }
}

// ---------------------------------------------------------------------------
// Node MLP: hn = LN(GELU([h|sum|max|mean|u] W1 + b1) W2 + b2)*g + be (+h)
// 32 threads = 1 wave per 16-node tile.  The scalar u feature is applied as a
// rank-1 correction (u_i * W1[last_row]) outside the WMMA loop.
// ---------------------------------------------------------------------------
__global__ __launch_bounds__(32) void node_mlp_kernel(
    const float* __restrict__ h_in, int h_stride,
    const float* __restrict__ ssum, const float* __restrict__ smx,
    const float* __restrict__ scnt, const float* __restrict__ u,
    const int* __restrict__ batch, float* __restrict__ h_out,
    const float* __restrict__ W1, const float* __restrict__ W2,
    const float* __restrict__ b1, const float* __restrict__ b2,
    const float* __restrict__ gg, const float* __restrict__ be,
    int N, int nin, int Kw, int residual)
{
  __shared__ alignas(16) bf16 sW1t[64 * KN_MAX];
  __shared__ alignas(16) bf16 sW2t[64 * 64];
  __shared__ alignas(16) bf16 sX[16 * KN_MAX];
  __shared__ alignas(16) bf16 sH[16 * 64];
  __shared__ float sY[16 * 64];
  __shared__ float sb1[64], sb2[64], sg[64], sbe[64], sW1u[64], su[16];

  const int lane = threadIdx.x;
  const int Kc = nin + 192;           // features fed through WMMA (u excluded)
  const int i0 = blockIdx.x * 16;

  for (int i = lane; i < 64 * Kw; i += 32) {
    int n = i / Kw, k = i - n * Kw;
    sW1t[i] = (k < Kc) ? (bf16)W1[k * 64 + n] : (bf16)0.0f;
  }
  for (int i = lane; i < 64 * 64; i += 32) {
    int n = i >> 6, k = i & 63;
    sW2t[i] = (bf16)W2[k * 64 + n];
  }
  for (int i = lane; i < 64; i += 32) {
    sb1[i] = b1[i]; sb2[i] = b2[i]; sg[i] = gg[i]; sbe[i] = be[i];
    sW1u[i] = W1[(long)Kc * 64 + i];  // W1 row for the scalar u feature
  }
  if (lane < 16) {
    int node = i0 + lane;
    su[lane] = (node < N) ? u[batch[node]] : 0.0f;
  }

  for (int i = lane; i < 16 * Kw; i += 32) {
    int r = i / Kw, k = i - r * Kw;
    int node = i0 + r;
    float v = 0.0f;
    if (node < N) {
      if (k < nin)            v = h_in[(long)node * h_stride + k];
      else if (k < nin + 64)  v = ssum[(long)node * 64 + (k - nin)];
      else if (k < nin + 128) {
        float m = smx[(long)node * 64 + (k - nin - 64)];
        v = (m == -__builtin_inff()) ? 0.0f : m;
      } else if (k < nin + 192) {
        float s = ssum[(long)node * 64 + (k - nin - 128)];
        v = s / fmaxf(scnt[node], 1.0f);
      }
    }
    sX[i] = (bf16)v;
  }
  __syncthreads();

  v8f acc[4];
#pragma unroll
  for (int c = 0; c < 4; ++c)
#pragma unroll
    for (int v = 0; v < 8; ++v) acc[c][v] = 0.0f;
  wave_gemm16(sX, sW1t, Kw, lane, acc);

  {
    const int m = lane & 15, half = lane >> 4;
#pragma unroll
    for (int c = 0; c < 4; ++c)
#pragma unroll
      for (int v = 0; v < 8; ++v) {
        int rowM = v + 8 * half, colf = 16 * c + m;
        float x = acc[c][v] + sb1[colf] + su[rowM] * sW1u[colf];
        sH[rowM * 64 + colf] = (bf16)gelu_exact(x);
      }
  }
  __syncthreads();

  v8f acc2[4];
#pragma unroll
  for (int c = 0; c < 4; ++c)
#pragma unroll
    for (int v = 0; v < 8; ++v) acc2[c][v] = 0.0f;
  wave_gemm16(sH, sW2t, 64, lane, acc2);

  {
    const int m = lane & 15, half = lane >> 4;
#pragma unroll
    for (int c = 0; c < 4; ++c)
#pragma unroll
      for (int v = 0; v < 8; ++v)
        sY[(v + 8 * half) * 64 + 16 * c + m] = acc2[c][v] + sb2[16 * c + m];
  }
  __syncthreads();

  if (lane < 16) {
    int node = i0 + lane;
    if (node < N) {
      const float* y = &sY[lane * 64];
      float mean = 0.0f;
      for (int f = 0; f < 64; ++f) mean += y[f];
      mean *= (1.0f / 64.0f);
      float var = 0.0f;
      for (int f = 0; f < 64; ++f) { float d = y[f] - mean; var += d * d; }
      float inv = rsqrtf(var * (1.0f / 64.0f) + 1e-5f);
      float* o = h_out + (long)node * 64;
      for (int f = 0; f < 64; ++f) {
        float ln = (y[f] - mean) * inv * sg[f] + sbe[f];
        o[f] = residual ? ln + h_in[(long)node * h_stride + f] : ln;
      }
    }
  }
}

// ---------------------------------------------------------------------------
// Global pooling + head
// ---------------------------------------------------------------------------
__global__ void node_pool_kernel(const float* __restrict__ h,
                                 const int* __restrict__ batch,
                                 float* __restrict__ gsum, float* __restrict__ gmax,
                                 float* __restrict__ gcnt, int N) {
  long idx = (long)blockIdx.x * blockDim.x + threadIdx.x;
  long stride = (long)gridDim.x * blockDim.x;
  long tot = (long)N * 64;
  for (; idx < tot; idx += stride) {
    int i = (int)(idx >> 6), f = (int)(idx & 63);
    int g = batch[i];
    float v = h[idx];
    atomicAdd(&gsum[g * 64 + f], v);
    atomicMaxF(&gmax[g * 64 + f], v);
    if (f == 0) atomicAdd(&gcnt[g], 1.0f);
  }
}

__global__ __launch_bounds__(256) void head_kernel(
    const float* __restrict__ gsum, const float* __restrict__ gmax,
    const float* __restrict__ gcnt, const float* __restrict__ u,
    const float* __restrict__ W1, const float* __restrict__ b1,
    const float* __restrict__ W2, const float* __restrict__ b2,
    const float* __restrict__ W3, const float* __restrict__ b3,
    const float* __restrict__ W4, const float* __restrict__ b4,
    float* __restrict__ out)
{
  __shared__ float sz[8][193];
  __shared__ float s1[8][64];
  __shared__ float s2[8][64];
  __shared__ float s3[8][64];
  int t = threadIdx.x;
  for (int i = t; i < 8 * 193; i += 256) {
    int g = i / 193, k = i - g * 193;
    float cnt = fmaxf(gcnt[g], 1.0f);
    float v;
    if (k < 64)       v = gsum[g * 64 + k];
    else if (k < 128) v = gsum[g * 64 + (k - 64)] / cnt;
    else if (k < 192) { float m = gmax[g * 64 + (k - 128)];
                        v = (m == -__builtin_inff()) ? 0.0f : m; }
    else              v = u[g];
    sz[g][k] = v;
  }
  __syncthreads();
  for (int i = t; i < 512; i += 256) {
    int g = i >> 6, f = i & 63;
    float a = b1[f];
    for (int k = 0; k < 193; ++k) a = fmaf(sz[g][k], W1[k * 64 + f], a);
    s1[g][f] = gelu_exact(a);
  }
  __syncthreads();
  for (int i = t; i < 512; i += 256) {
    int g = i >> 6, f = i & 63;
    float a = b2[f];
    for (int k = 0; k < 64; ++k) a = fmaf(s1[g][k], W2[k * 64 + f], a);
    s2[g][f] = gelu_exact(a);
  }
  __syncthreads();
  for (int i = t; i < 512; i += 256) {
    int g = i >> 6, f = i & 63;
    float a = b3[f];
    for (int k = 0; k < 64; ++k) a = fmaf(s2[g][k], W3[k * 64 + f], a);
    s3[g][f] = gelu_exact(a);
  }
  __syncthreads();
  if (t < 16) {
    int g = t >> 1, o = t & 1;
    float a = b4[o];
    for (int k = 0; k < 64; ++k) a = fmaf(s3[g][k], W4[k * 2 + o], a);
    float sp = (a > 20.0f) ? a : log1pf(expf(a));
    out[g * 2 + o] = sp * ((o == 1) ? 0.85f : 1.0f);
  }
}

__global__ void fill_f32(float* __restrict__ p, long n, float v) {
  long i = (long)blockIdx.x * blockDim.x + threadIdx.x;
  long stride = (long)gridDim.x * blockDim.x;
  for (; i < n; i += stride) p[i] = v;
}

// ---------------------------------------------------------------------------
// Host orchestration.
// Input order: x, edge_index, edge_attr, u, batch, then 44 param leaves in JAX
// tree order: per block sorted keys = W1,W2,b1,b2,be,g; layers[0..2]{edge,node},
// then out{W1..W4,b1..b4}.
// ---------------------------------------------------------------------------
extern "C" void kernel_launch(void* const* d_in, const int* in_sizes, int n_in,
                              void* d_out, int out_size, void* d_ws, size_t ws_size,
                              hipStream_t stream) {
  const float* x    = (const float*)d_in[0];
  const int*   ei   = (const int*)d_in[1];
  const float* ea0  = (const float*)d_in[2];
  const float* u    = (const float*)d_in[3];
  const int*   batch= (const int*)d_in[4];
  const int N = in_sizes[4];
  const int E = in_sizes[1] / 2;
  const int* row = ei;
  const int* col = ei + E;

  const float* P[44];
  for (int i = 0; i < 44 && (5 + i) < n_in; ++i) P[i] = (const float*)d_in[5 + i];

  float* ws   = (float*)d_ws;
  float* ea64 = ws;                                 // [E,64]
  float* h64  = ea64 + (size_t)E * 64;              // [N,64]
  float* ssum = h64  + (size_t)N * 64;              // [N,64]
  float* smx  = ssum + (size_t)N * 64;              // [N,64]
  float* scnt = smx  + (size_t)N * 64;              // [N]
  float* gsum = scnt + (size_t)N;                   // [8,64]
  float* gmax = gsum + 512;                         // [8,64]
  float* gcnt = gmax + 512;                         // [8]

  const float NEG_INF = -__builtin_inff();
  const int sc_blocks = (int)(((long)E * 64 + 255) / 256);

  for (int l = 0; l < 3; ++l) {
    const float* const* eb = &P[12 * l];
    const float* const* nb = &P[12 * l + 6];
    const int nin  = l ? 64 : 6;
    const int ein  = l ? 64 : 3;
    const int Kin  = 2 * nin + ein;                 // 15 or 192
    const int Kpad = (Kin + 31) & ~31;              // 32 or 192
    const float* hcur = l ? h64 : x;   const int hs  = l ? 64 : 6;
    const float* eain = l ? ea64 : ea0; const int eas = l ? 64 : 3;

    edge_mlp_kernel<<<(E + 31) / 32, 64, 0, stream>>>(
        hcur, hs, row, col, eain, eas, ea64,
        eb[0], eb[1], eb[2], eb[3], eb[5], eb[4],   // W1,W2,b1,b2,g,be
        E, nin, Kin, Kpad, l > 0);

    fill_f32<<<(int)(((long)N * 64 + 255) / 256), 256, 0, stream>>>(ssum, (long)N * 64, 0.0f);
    fill_f32<<<(int)(((long)N * 64 + 255) / 256), 256, 0, stream>>>(smx,  (long)N * 64, NEG_INF);
    fill_f32<<<(N + 255) / 256, 256, 0, stream>>>(scnt, N, 0.0f);
    edge_scatter_kernel<<<sc_blocks, 256, 0, stream>>>(ea64, col, ssum, smx, scnt, E);

    const int Kw = ((nin + 192) + 31) & ~31;        // 224 or 256
    node_mlp_kernel<<<(N + 15) / 16, 32, 0, stream>>>(
        hcur, hs, ssum, smx, scnt, u, batch, h64,
        nb[0], nb[1], nb[2], nb[3], nb[5], nb[4],   // W1,W2,b1,b2,g,be
        N, nin, Kw, l > 0);
  }

  fill_f32<<<2, 256, 0, stream>>>(gsum, 512, 0.0f);
  fill_f32<<<2, 256, 0, stream>>>(gmax, 512, NEG_INF);
  fill_f32<<<1, 256, 0, stream>>>(gcnt, 8, 0.0f);
  node_pool_kernel<<<(int)(((long)N * 64 + 255) / 256), 256, 0, stream>>>(
      h64, batch, gsum, gmax, gcnt, N);

  const float* const* ob = &P[36];                  // W1,W2,W3,W4,b1,b2,b3,b4
  head_kernel<<<1, 256, 0, stream>>>(gsum, gmax, gcnt, u,
      ob[0], ob[4], ob[1], ob[5], ob[2], ob[6], ob[3], ob[7],
      (float*)d_out);
}
